// BiLstmCRF_75617194213473
// MI455X (gfx1250) — compile-verified, matrix operand order checked by reading, not run on hardware
//
#include <hip/hip_runtime.h>
#include <stdint.h>

// ---------------- CDNA5 WMMA types ----------------
typedef __attribute__((ext_vector_type(16))) __bf16 v16bf;
typedef __attribute__((ext_vector_type(8)))  float  v8f;

union FragAB { uint32_t u[8]; v16bf v; };
union FragC  { float    f[8]; v8f   v; };

__device__ __forceinline__ uint16_t f2bf(float x) {
  uint32_t u = __builtin_bit_cast(uint32_t, x);
  u += 0x7FFFu + ((u >> 16) & 1u);          // round-to-nearest-even
  return (uint16_t)(u >> 16);
}
__device__ __forceinline__ float sigmoidf_(float x) { return 1.0f / (1.0f + __expf(-x)); }
__device__ __forceinline__ float tanhf_(float x) {   // hw v_exp_f32 based, clamped
  x = fminf(fmaxf(x, -15.f), 15.f);
  float e = __expf(2.f * x);
  return (e - 1.f) / (e + 1.f);
}

// K-pair (2 bf16 = 1 dword) offset within a 32-wide K slice for 16-bit WMMA
// fragments (ISA 7.12.2): VGPR v<4 -> K=2v ; v>=4 -> K=2v+8 ; upper lane half +8.
__device__ __forceinline__ int kpair(int v, int lane) {
  int k = (v < 4) ? (2 * v) : (2 * v + 8);
  return k + ((lane >> 4) << 3);
}

// ---- split workgroup barrier (gfx12/CDNA5): signal, overlap, wait ----
#if defined(__has_builtin)
#if __has_builtin(__builtin_amdgcn_s_barrier_signal) && __has_builtin(__builtin_amdgcn_s_barrier_wait)
#define CDNA5_SPLIT_BARRIER 1
#endif
#endif

__device__ __forceinline__ void lds_writes_visible() {
  asm volatile("s_wait_dscnt 0x0" ::: "memory");
}
__device__ __forceinline__ void bar_signal() {
#ifdef CDNA5_SPLIT_BARRIER
  lds_writes_visible();
  __builtin_amdgcn_s_barrier_signal(-1);
#else
  __syncthreads();
#endif
}
__device__ __forceinline__ void bar_wait() {
#ifdef CDNA5_SPLIT_BARRIER
  __builtin_amdgcn_s_barrier_wait(-1);
#endif
}

// ---------------- embedding gather -> bf16, time-major ----------------
__global__ void gather_embed_kernel(const int* __restrict__ inputs,  // [B][T]
                                    const float* __restrict__ emb,   // [V][128]
                                    uint16_t* __restrict__ xT,       // [T][B][128]
                                    int T, int B) {
  size_t i = (size_t)blockIdx.x * blockDim.x + threadIdx.x;
  size_t total = (size_t)T * B * 128;
  if (i >= total) return;
  int d = (int)(i & 127);
  int b = (int)((i >> 7) % B);
  int t = (int)(i / ((size_t)B * 128));
  int tok = inputs[(size_t)b * T + t];
  xT[i] = f2bf(emb[(size_t)tok * 128 + d]);
}

// ---------------- fp32 -> bf16 weight convert ----------------
__global__ void cvt_bf16_kernel(const float* __restrict__ src, uint16_t* __restrict__ dst, int n) {
  int i = blockIdx.x * blockDim.x + threadIdx.x;
  if (i < n) dst[i] = f2bf(src[i]);
}

// ---------------- one BiLSTM layer (both directions via blockIdx.y) ----------------
// Grid: (8 row-groups of 16 batch rows, 2 directions), 256 threads = 8 waves.
// Critical recurrence h@Whh: Whh fragments live in REGISTERS (16 frags/wave),
// Wih staged in LDS (CDNA5 320KB WGP LDS), x streamed from L2 with prefetch.
// Split barrier pipelines x_{t+1}@Wih into the h_t-exchange barrier shadow.
template <int DIN>
__global__ __launch_bounds__(256)
void lstm_layer_kernel(const uint16_t* __restrict__ X,       // [T][B][DIN] bf16
                       const uint16_t* __restrict__ WihAll,  // [2][512][DIN] bf16
                       const uint16_t* __restrict__ WhhAll,  // [2][512][128] bf16
                       const float*    __restrict__ biasAll, // [2][512]
                       uint16_t* __restrict__ Hout,          // [T][B][256] bf16
                       int T, int B) {
  extern __shared__ uint16_t smem[];
  uint16_t* wih  = smem;              // 512*DIN bf16
  uint16_t* hbuf = smem + 512 * DIN;  // 16*128 bf16

  const int dir     = blockIdx.y;
  const int rowBase = blockIdx.x * 16;
  const int tid     = threadIdx.x;
  const int lane    = tid & 31;
  const int wn      = tid >> 5;   // wave id 0..7 -> hidden-column tile
  const int nlo     = lane & 15;

  const uint16_t* WihG = WihAll + (size_t)dir * 512 * DIN;
  const uint16_t* WhhG = WhhAll + (size_t)dir * 512 * 128;
  const float*    bias = biasAll + dir * 512;
  const int colOff     = dir * 128;

  // Stage Wih into LDS
  {
    const uint4* s = (const uint4*)WihG;
    uint4* d = (uint4*)wih;
    for (int i = tid; i < 512 * DIN / 8; i += 256) d[i] = s[i];
  }
  // h0 = 0
  {
    uint32_t* hz = (uint32_t*)hbuf;
    for (int i = tid; i < 16 * 128 / 2; i += 256) hz[i] = 0u;
  }

  // Whh fragments -> registers: this wave's 16 gate-columns, all 4 gates, 4 K-slices
  FragAB whhR[4][4];
  #pragma unroll
  for (int g = 0; g < 4; ++g)
    #pragma unroll
    for (int kk = 0; kk < 4; ++kk) {
      const uint16_t* pb = WhhG + (size_t)(g * 128 + wn * 16 + nlo) * 128 + kk * 32;
      #pragma unroll
      for (int v = 0; v < 8; ++v) whhR[g][kk].u[v] = *(const uint32_t*)(pb + kpair(v, lane));
    }

  float bi[4];
  #pragma unroll
  for (int g = 0; g < 4; ++g) bi[g] = bias[g * 128 + wn * 16 + nlo];

  FragC cst;
  #pragma unroll
  for (int r = 0; r < 8; ++r) cst.f[r] = 0.0f;

  __syncthreads();   // wih + hbuf staged

  auto tOf = [&](int step) { return dir ? (T - 1 - step) : step; };

  // x_t @ Wih^T : A from global (prefetched), B from LDS. Independent of h.
  auto xpart = [&](FragC* acc, int step) {
    const int t = tOf(step);
    #pragma unroll
    for (int g = 0; g < 4; ++g)
      #pragma unroll
      for (int r = 0; r < 8; ++r) acc[g].f[r] = bi[g];
    const uint16_t* xrow = X + ((size_t)t * B + rowBase + nlo) * DIN;
    #pragma unroll
    for (int kk = 0; kk < DIN / 32; ++kk) {
      FragAB a;
      #pragma unroll
      for (int v = 0; v < 8; ++v) a.u[v] = *(const uint32_t*)(xrow + kk * 32 + kpair(v, lane));
      #pragma unroll
      for (int g = 0; g < 4; ++g) {
        FragAB bf;
        const uint16_t* pb = wih + (g * 128 + wn * 16 + nlo) * DIN + kk * 32;
        #pragma unroll
        for (int v = 0; v < 8; ++v) bf.u[v] = *(const uint32_t*)(pb + kpair(v, lane));
        acc[g].v = __builtin_amdgcn_wmma_f32_16x16x32_bf16(
            false, a.v, false, bf.v, (short)0, acc[g].v, false, false);
      }
    }
    if (step + 1 < T) {  // warm next activations (global_prefetch_b8)
      __builtin_prefetch(X + ((size_t)tOf(step + 1) * B + rowBase + nlo) * DIN, 0, 0);
    }
  };

  // h_{t-1} @ Whh^T (register B operands), then gates/state/h-writeback.
  auto hpart_gates = [&](FragC* acc, int step) {
    const int t = tOf(step);
    #pragma unroll
    for (int kk = 0; kk < 4; ++kk) {
      FragAB a;
      const uint16_t* pa = hbuf + nlo * 128 + kk * 32;
      #pragma unroll
      for (int v = 0; v < 8; ++v) a.u[v] = *(const uint32_t*)(pa + kpair(v, lane));
      #pragma unroll
      for (int g = 0; g < 4; ++g)
        acc[g].v = __builtin_amdgcn_wmma_f32_16x16x32_bf16(
            false, a.v, false, whhR[g][kk].v, (short)0, acc[g].v, false, false);
    }
    #pragma unroll
    for (int r = 0; r < 8; ++r) {
      float zi = acc[0].f[r], zf = acc[1].f[r], zg = acc[2].f[r], zo = acc[3].f[r];
      float c = sigmoidf_(zf) * cst.f[r] + sigmoidf_(zi) * tanhf_(zg);
      float h = sigmoidf_(zo) * tanhf_(c);
      cst.f[r] = c;
      int m   = ((lane >> 4) << 3) + r;   // C-tile row mapping
      int col = wn * 16 + nlo;
      uint16_t hb = f2bf(h);
      hbuf[m * 128 + col] = hb;
      Hout[((size_t)t * B + rowBase + m) * 256 + colOff + col] = hb;
    }
  };

  FragC accA[4], accB[4];
  xpart(accA, 0);                       // prologue: input-side of step 0

  for (int s = 0; s < T; s += 2) {
    // ---- step s ----
    hpart_gates(accA, s);
    bar_signal();                       // h_s written (s_wait_dscnt + signal)
    if (s + 1 < T) xpart(accB, s + 1);  // overlapped in barrier shadow
    bar_wait();                         // h_s visible to all waves
    // ---- step s+1 ----
    hpart_gates(accB, s + 1);
    bar_signal();
    if (s + 2 < T) xpart(accA, s + 2);
    bar_wait();
  }
}

// ---------------- emissions: [T*B,256] @ Wout^T + b_out -> [T*B,32] ----------------
__global__ __launch_bounds__(256)
void emission_kernel(const uint16_t* __restrict__ H2,    // [TB][256] bf16
                     const uint16_t* __restrict__ Wout,  // [32][256] bf16
                     const float* __restrict__ bout,     // [32]
                     float* __restrict__ em,             // [TB][32]
                     int TB) {
  int tid = threadIdx.x, lane = tid & 31, w = tid >> 5;
  int mt = w >> 1, nt = w & 1;
  int nlo = lane & 15;
  size_t rowBase = (size_t)blockIdx.x * 64 + (size_t)mt * 16;
  int jcol = nt * 16 + nlo;

  FragC acc;
  float bv = bout[jcol];
  #pragma unroll
  for (int r = 0; r < 8; ++r) acc.f[r] = bv;

  const uint16_t* arow = H2 + (rowBase + nlo) * 256;
  #pragma unroll
  for (int kk = 0; kk < 8; ++kk) {
    FragAB a, b;
    const uint16_t* pb = Wout + jcol * 256 + kk * 32;
    #pragma unroll
    for (int v = 0; v < 8; ++v) {
      int k = kpair(v, lane);
      a.u[v] = *(const uint32_t*)(arow + kk * 32 + k);
      b.u[v] = *(const uint32_t*)(pb + k);
    }
    acc.v = __builtin_amdgcn_wmma_f32_16x16x32_bf16(
        false, a.v, false, b.v, (short)0, acc.v, false, false);
  }
  #pragma unroll
  for (int r = 0; r < 8; ++r) {
    int m = ((lane >> 4) << 3) + r;
    em[(rowBase + m) * 32 + jcol] = acc.f[r];
  }
}

// ---------------- Viterbi: forward + in-block backtrace. K=32 == wave32 ----------------
__global__ __launch_bounds__(128)
void viterbi_kernel(const float* __restrict__ em,       // [T][B][32]
                    const float* __restrict__ start_t,  // [32]
                    const float* __restrict__ end_t,    // [32]
                    const float* __restrict__ trans,    // [32][32]
                    uint8_t* __restrict__ hist,         // [T][B][32]
                    int* __restrict__ out,              // [B][T]
                    int T, int B) {
  __shared__ float transLDS[32 * 32];
  __shared__ float sLDS[128];

  int tid = threadIdx.x;
  int j  = tid & 31;
  int bb = tid >> 5;                       // 0..3 local batch
  int b  = blockIdx.x * 4 + bb;

  for (int i = tid; i < 1024; i += 128) transLDS[i] = trans[i];
  __syncthreads();

  float s = start_t[j] + em[(size_t)b * 32 + j];

  for (int t = 1; t < T; ++t) {
    sLDS[tid] = s;
    __syncthreads();
    const float* sb = &sLDS[bb * 32];
    float best = -3.0e38f; int bi = 0;
    #pragma unroll
    for (int i = 0; i < 32; ++i) {
      float v = sb[i] + transLDS[i * 32 + j];
      if (v > best) { best = v; bi = i; }
    }
    s = best + em[((size_t)t * B + b) * 32 + j];
    hist[((size_t)t * B + b) * 32 + j] = (uint8_t)bi;
    __syncthreads();
  }

  s += end_t[j];
  sLDS[tid] = s;
  __threadfence_block();
  __syncthreads();
  if (j == 0) {
    float best = sLDS[bb * 32]; int bi = 0;
    for (int i = 1; i < 32; ++i) {
      float v = sLDS[bb * 32 + i];
      if (v > best) { best = v; bi = i; }
    }
    int tag = bi;
    out[(size_t)b * T + (T - 1)] = tag;
    for (int t = T - 1; t >= 1; --t) {
      tag = hist[((size_t)t * B + b) * 32 + tag];
      out[(size_t)b * T + (t - 1)] = tag;
    }
  }
}

// ---------------- host orchestration ----------------
extern "C" void kernel_launch(void* const* d_in, const int* in_sizes, int n_in,
                              void* d_out, int out_size, void* d_ws, size_t ws_size,
                              hipStream_t stream) {
  const int B = 128, T = 512, D = 128;
  const int TB = T * B;

  const int*   inputs  = (const int*)d_in[0];
  // d_in[1] = tags (training only, unused)
  const float* emb     = (const float*)d_in[2];
  const float* w_ih_l0 = (const float*)d_in[3];
  const float* w_hh_l0 = (const float*)d_in[4];
  const float* b_l0    = (const float*)d_in[5];
  const float* w_ih_l1 = (const float*)d_in[6];
  const float* w_hh_l1 = (const float*)d_in[7];
  const float* b_l1    = (const float*)d_in[8];
  const float* W_out   = (const float*)d_in[9];
  const float* b_out   = (const float*)d_in[10];
  const float* start_t = (const float*)d_in[11];
  const float* end_t   = (const float*)d_in[12];
  const float* trans   = (const float*)d_in[13];
  (void)in_sizes; (void)n_in; (void)out_size; (void)ws_size;

  char* ws = (char*)d_ws;
  size_t off = 0;
  auto wsalloc = [&](size_t bytes) -> void* {
    void* p = ws + off;
    off += (bytes + 255) & ~(size_t)255;
    return p;
  };
  uint16_t* xT    = (uint16_t*)wsalloc((size_t)TB * D * 2);          // 16 MB
  uint16_t* wih0  = (uint16_t*)wsalloc((size_t)2 * 512 * 128 * 2);
  uint16_t* whh0  = (uint16_t*)wsalloc((size_t)2 * 512 * 128 * 2);
  uint16_t* wih1  = (uint16_t*)wsalloc((size_t)2 * 512 * 256 * 2);
  uint16_t* whh1  = (uint16_t*)wsalloc((size_t)2 * 512 * 128 * 2);
  uint16_t* woutb = (uint16_t*)wsalloc((size_t)32 * 256 * 2);
  uint16_t* H1    = (uint16_t*)wsalloc((size_t)TB * 256 * 2);        // 32 MB
  uint16_t* H2    = (uint16_t*)wsalloc((size_t)TB * 256 * 2);        // 32 MB
  float*    em    = (float*)   wsalloc((size_t)TB * 32 * 4);         // 8 MB
  uint8_t*  hist  = (uint8_t*) wsalloc((size_t)TB * 32);             // 2 MB

  // 1) embedding gather (fp32 -> bf16, time-major)
  {
    size_t total = (size_t)TB * D;
    gather_embed_kernel<<<(unsigned)((total + 255) / 256), 256, 0, stream>>>(
        inputs, emb, xT, T, B);
  }
  // 2) weight conversion to bf16
  auto cvt = [&](const float* s, uint16_t* d, int n) {
    cvt_bf16_kernel<<<(n + 255) / 256, 256, 0, stream>>>(s, d, n);
  };
  cvt(w_ih_l0, wih0, 2 * 512 * 128);
  cvt(w_hh_l0, whh0, 2 * 512 * 128);
  cvt(w_ih_l1, wih1, 2 * 512 * 256);
  cvt(w_hh_l1, whh1, 2 * 512 * 128);
  cvt(W_out,  woutb, 32 * 256);

  // 3) BiLSTM layers: Wih + h in dynamic LDS (CDNA5 320KB WGP LDS), Whh in VGPRs
  lstm_layer_kernel<128><<<dim3(8, 2), 256, (size_t)(512 * 128 + 16 * 128) * 2, stream>>>(
      xT, wih0, whh0, b_l0, H1, T, B);
  lstm_layer_kernel<256><<<dim3(8, 2), 256, (size_t)(512 * 256 + 16 * 128) * 2, stream>>>(
      H1, wih1, whh1, b_l1, H2, T, B);

  // 4) emissions (WMMA GEMM, N=32)
  emission_kernel<<<TB / 64, 256, 0, stream>>>(H2, woutb, b_out, em, TB);

  // 5) Viterbi forward + backtrace
  viterbi_kernel<<<B / 4, 128, 0, stream>>>(
      em, start_t, end_t, trans, hist, (int*)d_out, T, B);
}
